// CaptioningRNN_74749610819709
// MI455X (gfx1250) — compile-verified, matrix operand order checked by reading, not run on hardware
//
#include <hip/hip_runtime.h>
#include <hip/hip_bf16.h>
#include <math.h>

// Problem constants (from reference): N=128, T=512, D=512, H=512, DA*DA=16
#define N_      128
#define T_      512
#define D_      512
#define H_      512
#define DA2     16
#define FOURH   2048
#define NWG     16          // persistent workgroups (guaranteed co-resident)
#define NTHREADS 256        // 8 wave32s

typedef __bf16 bf16_t;
typedef __attribute__((ext_vector_type(16))) __bf16 v16bf;
typedef __attribute__((ext_vector_type(8)))  float  v8f;

// ---------------------------------------------------------------------------
// Prep kernels
// ---------------------------------------------------------------------------

__global__ void init_bar_kernel(unsigned* bar) {
    if (threadIdx.x == 0 && blockIdx.x == 0) *bar = 0u;
}

// fp32 -> bf16 conversion (grid-stride).
__global__ void cvt_bf16_kernel(const float* __restrict__ src,
                                bf16_t* __restrict__ dst, int total) {
    for (int i = blockIdx.x * blockDim.x + threadIdx.x; i < total;
         i += gridDim.x * blockDim.x)
        dst[i] = (bf16_t)src[i];
}

// Pack a (K=512, 4H=2048) fp32 weight into per-fragment-contiguous layout:
//   P[(kb*128 + nt)*512 + L*16 + kk] = W[(kb*32 + (L>>4)*16 + kk)*FOURH + nt*16 + (L&15)]
// A whole 32(K)x16(N) WMMA B-fragment is 1 KB contiguous; lane L reads 16
// contiguous bf16 (32 B) at lane-constant offset L*32 B within the fragment.
__global__ void pack_w_kernel(const float* __restrict__ W,
                              bf16_t* __restrict__ P) {
    const int total = D_ * FOURH; // 1M
    for (int p = blockIdx.x * blockDim.x + threadIdx.x; p < total;
         p += gridDim.x * blockDim.x) {
        const int kk = p & 15;
        const int L  = (p >> 4) & 31;
        const int nt = (p >> 9) & 127;
        const int kb = p >> 16;
        P[p] = (bf16_t)W[(kb * 32 + (L >> 4) * 16 + kk) * FOURH + nt * 16 + (L & 15)];
    }
}

// ---------------------------------------------------------------------------
// Persistent recurrent kernel
// ---------------------------------------------------------------------------

__device__ __forceinline__ float sigmoidf_(float x) {
    return 1.0f / (1.0f + __expf(-x));
}

// Device-scope grid barrier over NWG co-resident workgroups (monotonic counter).
__device__ __forceinline__ void grid_barrier(unsigned* bar, unsigned target) {
    __syncthreads();
    if (threadIdx.x == 0) {
        __hip_atomic_fetch_add(bar, 1u, __ATOMIC_ACQ_REL,
                               __HIP_MEMORY_SCOPE_AGENT);
        while (__hip_atomic_load(bar, __ATOMIC_ACQUIRE,
                                 __HIP_MEMORY_SCOPE_AGENT) < target) {
            __builtin_amdgcn_s_sleep(2);
        }
    }
    __syncthreads();
}

__global__ void __launch_bounds__(NTHREADS, 1)
rnn_persistent_kernel(const float*  __restrict__ A,       // (N,H,16) fp32
                      const float*  __restrict__ bias,    // (4H) fp32
                      const bf16_t* __restrict__ xb,      // (N,T,D) bf16
                      const bf16_t* __restrict__ WxP,     // packed (global/L2)
                      const bf16_t* __restrict__ WhP,     // packed (-> LDS)
                      const bf16_t* __restrict__ WaP,     // packed (-> LDS)
                      float*  __restrict__ h_state,       // (N,H) fp32
                      float*  __restrict__ c_state,       // (N,H) fp32
                      bf16_t* __restrict__ h_bf,          // (N,H) bf16
                      bf16_t* __restrict__ attn_bf,       // (N,H) bf16
                      unsigned* __restrict__ bar,
                      float*  __restrict__ out)           // (N,T,H) fp32
{
    // 2 x 128 KB of recurrent weights resident in LDS for the whole kernel
    // (CDNA5 WGP has 320 KB LDS). Layout: [(kb*8 + ct)*512 + lane*16 + kk].
    __shared__ bf16_t ldsWh[16 * 8 * 512];
    __shared__ bf16_t ldsWa[16 * 8 * 512];

    const int wg    = blockIdx.x;       // 0..15
    const int tid   = threadIdx.x;
    const int wave  = tid >> 5;         // 0..7
    const int lane  = tid & 31;
    const int lhalf = lane >> 4;
    const int l16   = lane & 15;

    // ---- one-time LDS fill: this WG's Wh / Wattn fragments -----------------
    for (int c = tid; c < 4096; c += NTHREADS) {   // 4096 chunks of 32 B
        const int e   = c * 16;                    // element offset in LDS
        const int blk = e >> 9;                    // kb*8 + ct
        const int kb  = blk >> 3;
        const int ct  = blk & 7;
        const int rem = e & 511;
        const int nt  = (ct >> 1) * 32 + wg * 2 + (ct & 1);
        const size_t s = ((size_t)(kb * 128 + nt)) * 512 + rem;
        *(v16bf*)(ldsWh + e) = *(const v16bf*)(WhP + s);
        *(v16bf*)(ldsWa + e) = *(const v16bf*)(WaP + s);
    }

    unsigned epoch = 0;

    // ---- h0 = c0 = mean over the 16 spatial locations of A (wave per n) ----
    {
        const int n = wg * 8 + wave;
        for (int hh = lane; hh < H_; hh += 32) {
            const float* ap = A + ((size_t)n * H_ + hh) * DA2;
            float s = 0.f;
            #pragma unroll
            for (int m = 0; m < 16; ++m) s += ap[m];
            const float h0 = s * (1.0f / 16.0f);
            h_state[(size_t)n * H_ + hh] = h0;
            c_state[(size_t)n * H_ + hh] = h0;
            h_bf  [(size_t)n * H_ + hh] = (bf16_t)h0;
        }
    }
    grid_barrier(bar, ++epoch * NWG);

    const float inv_sqrt_h = 0.04419417382415922f; // 1/sqrt(512)
    const int j0 = wg * 32;      // this WG's 32 hidden columns
    const int m0 = wave * 16;    // this wave's 16 batch rows
    const int arow = m0 + l16;   // A-fragment row for this lane

    // Loop-invariant bases (all hot-loop loads get compile-time immediates).
    const bf16_t* wx_lane = WxP + (size_t)wg * 1024 + (size_t)lane * 16;
    const bf16_t* lh_lane = ldsWh + lane * 16;
    const bf16_t* la_lane = ldsWa + lane * 16;
    const bf16_t* xrow    = xb + ((size_t)arow * T_) * D_ + lhalf * 16;
    const bf16_t* hrow    = h_bf + (size_t)arow * H_ + lhalf * 16;
    const bf16_t* atrow   = attn_bf + (size_t)arow * H_ + lhalf * 16;

    // Bias per accumulator tile (constant across time) hoisted out of the loop.
    float bv[8];
    #pragma unroll
    for (int ct = 0; ct < 8; ++ct)
        bv[ct] = bias[(ct >> 1) * H_ + j0 + (ct & 1) * 16 + l16];

    for (int t = 0; t < T_; ++t) {
        // ================= Phase A: attention (one wave per batch n) ========
        {
            const int n = wg * 8 + wave;
            float sc[16];
            #pragma unroll
            for (int m = 0; m < 16; ++m) sc[m] = 0.f;
            for (int hh = lane; hh < H_; hh += 32) {
                const float hv = h_state[(size_t)n * H_ + hh];
                const float* ap = A + ((size_t)n * H_ + hh) * DA2;
                #pragma unroll
                for (int m = 0; m < 16; ++m) sc[m] = fmaf(hv, ap[m], sc[m]);
            }
            #pragma unroll
            for (int off = 16; off >= 1; off >>= 1) {
                #pragma unroll
                for (int m = 0; m < 16; ++m)
                    sc[m] += __shfl_xor(sc[m], off, 32);
            }
            float mx = -3.0e38f;
            #pragma unroll
            for (int m = 0; m < 16; ++m) {
                sc[m] *= inv_sqrt_h;
                mx = fmaxf(mx, sc[m]);
            }
            float ssum = 0.f;
            #pragma unroll
            for (int m = 0; m < 16; ++m) { sc[m] = __expf(sc[m] - mx); ssum += sc[m]; }
            const float rs = 1.0f / ssum;
            #pragma unroll
            for (int m = 0; m < 16; ++m) sc[m] *= rs;
            for (int hh = lane; hh < H_; hh += 32) {
                const float* ap = A + ((size_t)n * H_ + hh) * DA2;
                float at = 0.f;
                #pragma unroll
                for (int m = 0; m < 16; ++m) at = fmaf(ap[m], sc[m], at);
                attn_bf[(size_t)n * H_ + hh] = (bf16_t)at;
            }
        }
        grid_barrier(bar, ++epoch * NWG);

        // ================= Phase B: fused 3-segment GEMM + LSTM gates =======
        v8f acc[8];
        #pragma unroll
        for (int ct = 0; ct < 8; ++ct) {
            #pragma unroll
            for (int r = 0; r < 8; ++r) acc[ct][r] = bv[ct];
        }

        // Segment 1: xt @ Wx (B fragments from L2, immediate-offset loads)
        {
            const bf16_t* abase = xrow + (size_t)t * D_;
            if (t + 1 < T_) __builtin_prefetch(abase + D_, 0, 1);
            #pragma unroll 4
            for (int kb = 0; kb < 16; ++kb) {
                const v16bf afrag = *(const v16bf*)(abase + kb * 32);
                #pragma unroll
                for (int ct = 0; ct < 8; ++ct) {
                    const int off = (kb * 128 + (ct >> 1) * 32 + (ct & 1)) * 512;
                    const v16bf bfrag = *(const v16bf*)(wx_lane + off);
                    acc[ct] = __builtin_amdgcn_wmma_f32_16x16x32_bf16(
                        false, afrag, false, bfrag, (short)0, acc[ct],
                        false, false);
                }
            }
        }
        // Segment 2: h @ Wh (B fragments from LDS)
        {
            #pragma unroll 4
            for (int kb = 0; kb < 16; ++kb) {
                const v16bf afrag = *(const v16bf*)(hrow + kb * 32);
                #pragma unroll
                for (int ct = 0; ct < 8; ++ct) {
                    const v16bf bfrag = *(const v16bf*)(lh_lane + (kb * 8 + ct) * 512);
                    acc[ct] = __builtin_amdgcn_wmma_f32_16x16x32_bf16(
                        false, afrag, false, bfrag, (short)0, acc[ct],
                        false, false);
                }
            }
        }
        // Segment 3: attn @ Wattn (B fragments from LDS)
        {
            #pragma unroll 4
            for (int kb = 0; kb < 16; ++kb) {
                const v16bf afrag = *(const v16bf*)(atrow + kb * 32);
                #pragma unroll
                for (int ct = 0; ct < 8; ++ct) {
                    const v16bf bfrag = *(const v16bf*)(la_lane + (kb * 8 + ct) * 512);
                    acc[ct] = __builtin_amdgcn_wmma_f32_16x16x32_bf16(
                        false, afrag, false, bfrag, (short)0, acc[ct],
                        false, false);
                }
            }
        }

        // LSTM pointwise update. C layout: vgpr r -> row m0 + r + 8*lhalf,
        // column = tile col base + l16.
        #pragma unroll
        for (int r = 0; r < 8; ++r) {
            const int n = m0 + r + lhalf * 8;
            #pragma unroll
            for (int s = 0; s < 2; ++s) {
                const int jj = j0 + s * 16 + l16;
                const float iv = acc[0 + s][r];
                const float fv = acc[2 + s][r];
                const float ov = acc[4 + s][r];
                const float gv = acc[6 + s][r];
                const size_t ci = (size_t)n * H_ + jj;
                const float cold = c_state[ci];
                const float cnew =
                    sigmoidf_(fv) * cold + sigmoidf_(iv) * tanhf(gv);
                const float hnew = sigmoidf_(ov) * tanhf(cnew);
                c_state[ci] = cnew;
                h_state[ci] = hnew;
                h_bf[ci]    = (bf16_t)hnew;
                out[((size_t)n * T_ + t) * H_ + jj] = hnew;
            }
        }
        grid_barrier(bar, ++epoch * NWG);
    }
}

// ---------------------------------------------------------------------------
// Host launcher
// ---------------------------------------------------------------------------

extern "C" void kernel_launch(void* const* d_in, const int* in_sizes, int n_in,
                              void* d_out, int out_size, void* d_ws, size_t ws_size,
                              hipStream_t stream) {
    const float* x     = (const float*)d_in[0];  // (N,T,D)
    const float* A     = (const float*)d_in[1];  // (N,H,4,4)
    const float* Wx    = (const float*)d_in[2];  // (D,4H)
    const float* Wh    = (const float*)d_in[3];  // (H,4H)
    const float* Wattn = (const float*)d_in[4];  // (H,4H)
    const float* b     = (const float*)d_in[5];  // (4H)
    float* out = (float*)d_out;                  // (N,T,H)

    char* ws = (char*)d_ws;
    size_t off = 0;
    auto carve = [&](size_t bytes) -> char* {
        char* p = ws + off;
        off = (off + bytes + 255) & ~(size_t)255;
        return p;
    };
    unsigned* bar    = (unsigned*)carve(256);
    bf16_t* xb       = (bf16_t*)carve((size_t)N_ * T_ * D_ * sizeof(bf16_t));
    bf16_t* WxP      = (bf16_t*)carve((size_t)D_ * FOURH * sizeof(bf16_t));
    bf16_t* WhP      = (bf16_t*)carve((size_t)H_ * FOURH * sizeof(bf16_t));
    bf16_t* WaP      = (bf16_t*)carve((size_t)H_ * FOURH * sizeof(bf16_t));
    float*  h_state  = (float*)carve((size_t)N_ * H_ * sizeof(float));
    float*  c_state  = (float*)carve((size_t)N_ * H_ * sizeof(float));
    bf16_t* h_bf     = (bf16_t*)carve((size_t)N_ * H_ * sizeof(bf16_t));
    bf16_t* attn_bf  = (bf16_t*)carve((size_t)N_ * H_ * sizeof(bf16_t));
    (void)ws_size; (void)in_sizes; (void)n_in; (void)out_size;

    init_bar_kernel<<<1, 64, 0, stream>>>(bar);
    cvt_bf16_kernel<<<8192, NTHREADS, 0, stream>>>(x, xb, N_ * T_ * D_);
    pack_w_kernel<<<2048, NTHREADS, 0, stream>>>(Wx,    WxP);
    pack_w_kernel<<<2048, NTHREADS, 0, stream>>>(Wh,    WhP);
    pack_w_kernel<<<2048, NTHREADS, 0, stream>>>(Wattn, WaP);
    rnn_persistent_kernel<<<NWG, NTHREADS, 0, stream>>>(
        A, b, xb, WxP, WhP, WaP, h_state, c_state, h_bf, attn_bf, bar, out);
}
